// Network_48275432407239
// MI455X (gfx1250) — matrix operand order repaired and checked by reading, not compiled, not run on hardware
//
#include <hip/hip_runtime.h>
#include <cmath>

// ---------------------------------------------------------------------------
// SLAYER SNN forward for MI455X (gfx1250, wave32, WMMA).
//   fc1 -> psp -> spike -> fc2 -> psp -> spike
// GEMMs run on v_wmma_f32_16x16x32_f16 with double-buffered LDS tiles;
// A-tile staging uses GLOBAL_LOAD_ASYNC_TO_LDS_B128 (ASYNCcnt) when the
// toolchain declares the builtin. PSP conv + spike scan are fused so each
// activation tensor is read exactly once.
// ---------------------------------------------------------------------------

typedef _Float16 v16h __attribute__((ext_vector_type(16)));
typedef _Float16 v8h  __attribute__((ext_vector_type(8)));
typedef float    v8f  __attribute__((ext_vector_type(8)));
typedef int      v4i  __attribute__((vector_size(4 * sizeof(int))));

#define GAS __attribute__((address_space(1)))
#define LAS __attribute__((address_space(3)))

#if defined(__HIP_DEVICE_COMPILE__) && \
    __has_builtin(__builtin_amdgcn_global_load_async_to_lds_b128) && \
    __has_builtin(__builtin_amdgcn_s_wait_asynccnt)
#define USE_ASYNC_LDS 1
#else
#define USE_ASYNC_LDS 0
#endif

#define BATCH   32
#define N_IN    1024
#define N_HID   2048
#define N_OUT   512
#define TSTEPS  256
#define KLEN    62      // alpha-kernel truncation length for tau=8, T=256
#define THETA   10.0f

// ---------------- f32 -> f16 weight conversion ------------------------------
__global__ void cvt_f32_f16_kernel(const float* __restrict__ in,
                                   _Float16* __restrict__ out, int n) {
  int i = blockIdx.x * blockDim.x + threadIdx.x;
  if (i < n) out[i] = (_Float16)in[i];
}

// ---------------- WMMA GEMM:  C[b] = W (MxK) * X[b] (KxT) -------------------
// Block: 256 threads (8 waves). Block tile: 128(M) x 64(N). Wave tile: 16x64.
// Double-buffered LDS; async DMA for the A tile, manual convert/transpose for B.
#define BLK_M 128
#define BLK_N 64
#define BLK_K 32
#define LDA_P 40   // padded LDS row stride (halves) for A: 80B, 16B-aligned
#define LDB_P 40   // padded LDS row stride (halves) for B^T

template <typename TB>
__global__ __launch_bounds__(256)
void wmma_gemm_kernel(const _Float16* __restrict__ W,  // M x K, row-major f16
                      const TB*       __restrict__ X,  // B x K x T
                      float*          __restrict__ C,  // B x M x T
                      int M, int K, int T) {
  __shared__ _Float16 ldsA[2][BLK_M * LDA_P];
  __shared__ _Float16 ldsB[2][BLK_N * LDB_P];   // transposed: [n][k]

  const int tid   = threadIdx.x;
  const int wave  = tid >> 5;      // 0..7 -> 16-row M stripe
  const int lane  = tid & 31;
  const int l15   = lane & 15;
  const int hi    = lane >> 4;     // half-wave select per WMMA layout
  const int b     = blockIdx.z;
  const int moBase = blockIdx.y * BLK_M;
  const int ntBase = blockIdx.x * BLK_N;

  const _Float16* Wb = W + (size_t)moBase * K;
  const TB*       Xb = X + (size_t)b * K * T + ntBase;

  v8f acc[4];
#pragma unroll
  for (int n = 0; n < 4; ++n)
#pragma unroll
    for (int j = 0; j < 8; ++j) acc[n][j] = 0.0f;

  const int ksteps = K / BLK_K;

  // Stage one 128x32 A tile + 32x64 B tile into LDS buffer `buf`.
  auto stage_tile = [&](int kt, int buf) {
    // --- A: straight 16B/lane copy -> async DMA to LDS when available ---
#pragma unroll
    for (int s = 0; s < 2; ++s) {
      int c8  = tid + 256 * s;          // 512 chunks of 8 halves
      int row = c8 >> 2;                // 4 chunks per 32-wide row
      int co  = (c8 & 3) * 8;
      const _Float16* gp = Wb + (size_t)row * K + (size_t)kt * BLK_K + co;
      _Float16* lp = &ldsA[buf][row * LDA_P + co];
#if USE_ASYNC_LDS
      __builtin_amdgcn_global_load_async_to_lds_b128(
          (GAS v4i*)(GAS void*)const_cast<_Float16*>(gp),
          (LAS v4i*)(LAS void*)lp, 0, 0);
#else
      *(v8h*)lp = *(const v8h*)gp;
#endif
    }
    // --- B: 32(k) x 64(n), converted/transposed into LDS as [n][k] ---
    if constexpr (sizeof(TB) == 4) {    // f32 spikes -> convert to f16
#pragma unroll
      for (int s = 0; s < 2; ++s) {
        int q   = tid + 256 * s;        // 512 float4 loads
        int row = q >> 4;               // k: 0..31
        int col = (q & 15) * 4;         // n
        const float4 f =
            *(const float4*)(const void*)(Xb + (size_t)(kt * BLK_K + row) * T + col);
        ldsB[buf][(col + 0) * LDB_P + row] = (_Float16)f.x;
        ldsB[buf][(col + 1) * LDB_P + row] = (_Float16)f.y;
        ldsB[buf][(col + 2) * LDB_P + row] = (_Float16)f.z;
        ldsB[buf][(col + 3) * LDB_P + row] = (_Float16)f.w;
      }
    } else {                            // already f16
      int q   = tid;                    // 256 chunks of 8 halves
      int row = q >> 3;                 // k: 0..31
      int col = (q & 7) * 8;            // n
      v8h h = *(const v8h*)(const void*)(Xb + (size_t)(kt * BLK_K + row) * T + col);
#pragma unroll
      for (int u = 0; u < 8; ++u) ldsB[buf][(col + u) * LDB_P + row] = h[u];
    }
  };

  stage_tile(0, 0);
#if USE_ASYNC_LDS
  __builtin_amdgcn_s_wait_asynccnt(0);
#endif
  __syncthreads();

  for (int kt = 0; kt < ksteps; ++kt) {
    const int cur = kt & 1;
    if (kt + 1 < ksteps) stage_tile(kt + 1, cur ^ 1);   // overlap with WMMAs
    if (kt + 2 < ksteps)                                 // warm B for kt+2
      __builtin_prefetch(Xb + (size_t)((kt + 2) * BLK_K + (tid & 31)) * T, 0, 0);

    // --- A fragment: lane holds row m = wave*16 + l15;
    //     halves 0..7 = K 8*hi..8*hi+7, halves 8..15 = K 16+8*hi.. ---
    const _Float16* pa = &ldsA[cur][(wave * 16 + l15) * LDA_P + 8 * hi];
    v8h alo = *(const v8h*)pa;
    v8h ahi = *(const v8h*)(pa + 16);
    v16h afrag = __builtin_shufflevector(alo, ahi, 0, 1, 2, 3, 4, 5, 6, 7,
                                         8, 9, 10, 11, 12, 13, 14, 15);
#pragma unroll
    for (int nsub = 0; nsub < 4; ++nsub) {
      // --- B fragment: lane holds col n; halves 0..7 = K 8*hi.., 8..15 = K 16+8*hi.. ---
      const _Float16* pb = &ldsB[cur][(nsub * 16 + l15) * LDB_P + 8 * hi];
      v8h blo = *(const v8h*)pb;
      v8h bhi = *(const v8h*)(pb + 16);
      v16h bfrag = __builtin_shufflevector(blo, bhi, 0, 1, 2, 3, 4, 5, 6, 7,
                                           8, 9, 10, 11, 12, 13, 14, 15);
      acc[nsub] = __builtin_amdgcn_wmma_f32_16x16x32_f16(
          false, afrag, false, bfrag, (short)0, acc[nsub], false, false);
    }

    if (kt + 1 < ksteps) {
#if USE_ASYNC_LDS
      __builtin_amdgcn_s_wait_asynccnt(0);
#endif
      __syncthreads();
    }
  }

  // --- epilogue: C layout VGPR r -> M = r + 8*hi, N = l15 ---
#pragma unroll
  for (int nsub = 0; nsub < 4; ++nsub)
#pragma unroll
    for (int r = 0; r < 8; ++r) {
      int row = moBase + wave * 16 + r + 8 * hi;
      int col = ntBase + nsub * 16 + l15;
      C[((size_t)b * M + row) * T + col] = acc[nsub][r];
    }
}

// ---------------- fused PSP conv + threshold/refractory scan ----------------
// One thread per (batch, channel). 64-slot LDS ring of past activations,
// spike history as 256-bit mask in registers.
#define PSPB 128

template <typename TOUT>
__global__ __launch_bounds__(PSPB)
void psp_spike_kernel(const float* __restrict__ A, TOUT* __restrict__ S,
                      int rows, int T) {
  __shared__ float srm_s[64];
  __shared__ float ref_s[64];
  __shared__ float ring[64][PSPB];

  const int tid = threadIdx.x;
  if (tid < 64) {   // build truncated alpha kernels (matches Python math.exp)
    double tt = (double)tid;
    double v  = (tt / 8.0) * exp(1.0 - tt / 8.0);
    srm_s[tid] = (tid < KLEN) ? (float)v : 0.0f;
    ref_s[tid] = (tid < KLEN) ? (float)(-2.0 * 10.0 * v) : 0.0f;
  }
  __syncthreads();

  const int gid = blockIdx.x * PSPB + tid;
  if (gid >= rows) return;
  const float* a = A + (size_t)gid * T;
  TOUT*        s = S + (size_t)gid * T;

  unsigned sbits[8] = {0u, 0u, 0u, 0u, 0u, 0u, 0u, 0u};

  for (int t = 0; t < T; ++t) {
    ring[t & 63][tid] = a[t];
    float u = 0.0f;                       // srm[0] == 0: a[t] contributes 0
    const int kmax = (t < KLEN - 1) ? t : (KLEN - 1);
    for (int k = 1; k <= kmax; ++k) {
      const int j = t - k;
      u += srm_s[k] * ring[j & 63][tid];                       // psp term
      const float sp = (float)((sbits[j >> 5] >> (j & 31)) & 1u);
      u += ref_s[k] * sp;                                      // refractory
    }
    const int fire = (u >= THETA);
    if (fire) sbits[t >> 5] |= (1u << (t & 31));
    s[t] = (TOUT)(fire ? 1.0f : 0.0f);    // /TS with TS=1
  }
}

// ---------------------------------------------------------------------------
extern "C" void kernel_launch(void* const* d_in, const int* in_sizes, int n_in,
                              void* d_out, int out_size, void* d_ws, size_t ws_size,
                              hipStream_t stream) {
  (void)in_sizes; (void)n_in; (void)out_size; (void)ws_size;
  const float* x  = (const float*)d_in[0];   // (32,1024,256)
  const float* w1 = (const float*)d_in[1];   // (2048,1024)
  const float* w2 = (const float*)d_in[2];   // (512,2048)
  float* out = (float*)d_out;                // (32,512,256)

  char* ws = (char*)d_ws;
  _Float16* w1h = (_Float16*)(ws);                         //  4 MB
  _Float16* w2h = (_Float16*)(ws + (size_t)4194304);       //  2 MB
  float*    a1  = (float*)   (ws + (size_t)6291456);       // 64 MB
  _Float16* s1h = (_Float16*)(ws + (size_t)73400320);      // 32 MB
  float*    a2  = (float*)   (ws + (size_t)106954752);     // 16 MB

  // weights -> f16
  cvt_f32_f16_kernel<<<(N_HID * N_IN + 255) / 256, 256, 0, stream>>>(
      w1, w1h, N_HID * N_IN);
  cvt_f32_f16_kernel<<<(N_OUT * N_HID + 255) / 256, 256, 0, stream>>>(
      w2, w2h, N_OUT * N_HID);

  // fc1: a1[b,o,t] = w1 @ x[b]
  wmma_gemm_kernel<float><<<dim3(TSTEPS / BLK_N, N_HID / BLK_M, BATCH),
                            256, 0, stream>>>(w1h, x, a1, N_HID, N_IN, TSTEPS);
  // psp + spike (layer 1) -> f16 spikes for next GEMM
  psp_spike_kernel<_Float16><<<(BATCH * N_HID) / PSPB, PSPB, 0, stream>>>(
      a1, s1h, BATCH * N_HID, TSTEPS);

  // fc2: a2[b,o,t] = w2 @ s1[b]
  wmma_gemm_kernel<_Float16><<<dim3(TSTEPS / BLK_N, N_OUT / BLK_M, BATCH),
                               256, 0, stream>>>(w2h, s1h, a2, N_OUT, N_HID, TSTEPS);
  // psp + spike (layer 2) -> f32 output
  psp_spike_kernel<float><<<(BATCH * N_OUT) / PSPB, PSPB, 0, stream>>>(
      a2, out, BATCH * N_OUT, TSTEPS);
}
